// StatefulPrefillLinearAttn_39376260170064
// MI455X (gfx1250) — matrix operand order repaired and checked by reading, not compiled, not run on hardware
//
#include <hip/hip_runtime.h>

// ---------------- problem constants ----------------
#define S_      4096
#define H_      16
#define DK_     128
#define DV_     128
#define Kc_     4
#define CS_     64
#define NC_     64
#define HID_    2048
#define KEY_DIM_  (H_*DK_)          // 2048
#define VAL_DIM_  (H_*DV_)          // 2048
#define CONV_DIM_ (2*KEY_DIM_+VAL_DIM_) // 6144

typedef __bf16 bf16_t;
typedef __attribute__((ext_vector_type(16))) __bf16 v16bf;
typedef __attribute__((ext_vector_type(8)))  float  v8f;

// ---------------- helpers ----------------
__device__ __forceinline__ bf16_t f2bf(float f) {
  unsigned u = __float_as_uint(f);
  u += 0x7FFFu + ((u >> 16) & 1u);          // round-to-nearest-even
  unsigned short h = (unsigned short)(u >> 16);
  return __builtin_bit_cast(__bf16, h);
}
__device__ __forceinline__ float siluf_(float x) { return x / (1.f + __expf(-x)); }
__device__ __forceinline__ float sigmoidf_(float x) { return 1.f / (1.f + __expf(-x)); }
__device__ __forceinline__ float softplusf_(float x) { return x > 20.f ? x : log1pf(__expf(x)); }

// ---------------- fp32 -> bf16 cast ----------------
__global__ void cast_bf16_kernel(const float* __restrict__ src, bf16_t* __restrict__ dst, int n) {
  int i = blockIdx.x * 256 + threadIdx.x;
  if (i < n) dst[i] = f2bf(src[i]);
}

// ---------------- WMMA bf16 GEMM: C[M,N] = A[M,K] * B[N,K]^T ----------------
__global__ __launch_bounds__(256) void gemm_bf16_kernel(
    const bf16_t* __restrict__ A, const bf16_t* __restrict__ B,
    float* __restrict__ C, int M, int N, int K) {
  const int lane = threadIdx.x & 31;
  const int wave = threadIdx.x >> 5;
  const int wm = wave >> 2, wn = wave & 3;
  const int mBase = blockIdx.y * 32 + wm * 16;
  const int nBase = blockIdx.x * 256 + wn * 64;
  const int r = lane & 15, half = lane >> 4;

  const bf16_t* pa = A + (size_t)(mBase + r) * K + half * 8;
  const bf16_t* pb0 = B + (size_t)(nBase +  0 + r) * K + half * 8;
  const bf16_t* pb1 = B + (size_t)(nBase + 16 + r) * K + half * 8;
  const bf16_t* pb2 = B + (size_t)(nBase + 32 + r) * K + half * 8;
  const bf16_t* pb3 = B + (size_t)(nBase + 48 + r) * K + half * 8;

  v8f acc0 = {}, acc1 = {}, acc2 = {}, acc3 = {};

  for (int k0 = 0; k0 < K; k0 += 32) {
    __builtin_prefetch((const void*)(pa + k0 + 256), 0, 1);
    __builtin_prefetch((const void*)(pb0 + k0 + 256), 0, 1);

    union { v16bf v; uint4 u[2]; } av;
    av.u[0] = *reinterpret_cast<const uint4*>(pa + k0);
    av.u[1] = *reinterpret_cast<const uint4*>(pa + k0 + 16);

    union { v16bf v; uint4 u[2]; } bv;
    bv.u[0] = *reinterpret_cast<const uint4*>(pb0 + k0);
    bv.u[1] = *reinterpret_cast<const uint4*>(pb0 + k0 + 16);
    acc0 = __builtin_amdgcn_wmma_f32_16x16x32_bf16(false, av.v, false, bv.v, (short)0, acc0, false, false);

    bv.u[0] = *reinterpret_cast<const uint4*>(pb1 + k0);
    bv.u[1] = *reinterpret_cast<const uint4*>(pb1 + k0 + 16);
    acc1 = __builtin_amdgcn_wmma_f32_16x16x32_bf16(false, av.v, false, bv.v, (short)0, acc1, false, false);

    bv.u[0] = *reinterpret_cast<const uint4*>(pb2 + k0);
    bv.u[1] = *reinterpret_cast<const uint4*>(pb2 + k0 + 16);
    acc2 = __builtin_amdgcn_wmma_f32_16x16x32_bf16(false, av.v, false, bv.v, (short)0, acc2, false, false);

    bv.u[0] = *reinterpret_cast<const uint4*>(pb3 + k0);
    bv.u[1] = *reinterpret_cast<const uint4*>(pb3 + k0 + 16);
    acc3 = __builtin_amdgcn_wmma_f32_16x16x32_bf16(false, av.v, false, bv.v, (short)0, acc3, false, false);
  }

  float* pc = C + (size_t)(mBase + 8 * half) * N + nBase + r;
#pragma unroll
  for (int e = 0; e < 8; ++e) {
    float* row = pc + (size_t)e * N;
    row[0]  = acc0[e];
    row[16] = acc1[e];
    row[32] = acc2[e];
    row[48] = acc3[e];
  }
}

// ---------------- tiny projections: b,a (N=16 each) ----------------
__global__ void ab_proj_kernel(const float* __restrict__ hidden,
                               const float* __restrict__ bw,
                               const float* __restrict__ aw,
                               float* __restrict__ ab) {
  int t = blockIdx.x * 128 + threadIdx.x;     // S*32
  if (t >= S_ * 32) return;
  int s = t >> 5, j = t & 31;
  const float* w = (j < 16) ? (bw + (size_t)j * HID_) : (aw + (size_t)(j - 16) * HID_);
  const float* hr = hidden + (size_t)s * HID_;
  float acc = 0.f;
  for (int k = 0; k < HID_; ++k) acc += hr[k] * w[k];
  ab[t] = acc;
}

// ---------------- depthwise causal conv (K=4) + SiLU ----------------
__global__ void conv_silu_kernel(const float* __restrict__ mixed,
                                 const float* __restrict__ cw,
                                 float* __restrict__ out) {
  int i = blockIdx.x * 256 + threadIdx.x;     // S*CONV_DIM
  if (i >= S_ * CONV_DIM_) return;
  int c = i % CONV_DIM_;
  int s = i / CONV_DIM_;
  float acc = 0.f;
#pragma unroll
  for (int j = 0; j < Kc_; ++j) {
    int ss = s - (Kc_ - 1) + j;
    if (ss >= 0) acc += cw[c * Kc_ + j] * mixed[(size_t)ss * CONV_DIM_ + c];
  }
  out[i] = siluf_(acc);
}

// conv_state[c][j] = mixed_qkv[S-K+j][c]  (pre-conv)
__global__ void conv_state_kernel(const float* __restrict__ mixed, float* __restrict__ out) {
  int t = blockIdx.x * 256 + threadIdx.x;     // CONV_DIM*K
  if (t >= CONV_DIM_ * Kc_) return;
  int c = t >> 2, j = t & 3;
  out[t] = mixed[(size_t)(S_ - Kc_ + j) * CONV_DIM_ + c];
}

// ---------------- beta / g gates ----------------
__global__ void gates_kernel(const float* __restrict__ ab, const float* __restrict__ Alog,
                             const float* __restrict__ dtb,
                             float* __restrict__ beta, float* __restrict__ g) {
  int t = blockIdx.x * 256 + threadIdx.x;     // S*H
  if (t >= S_ * H_) return;
  int s = t >> 4, h = t & 15;
  beta[t] = sigmoidf_(ab[s * 32 + h]);
  g[t] = -__expf(Alog[h]) * softplusf_(ab[s * 32 + 16 + h] + dtb[h]);
}

// per-(h,chunk) cumsum of g along s
__global__ void cumsum_g_kernel(const float* __restrict__ g, float* __restrict__ gc) {
  int t = blockIdx.x * 256 + threadIdx.x;     // H*NC
  if (t >= H_ * NC_) return;
  int h = t >> 6, nc = t & 63;
  float acc = 0.f;
  for (int i = 0; i < CS_; ++i) {
    int s = nc * CS_ + i;
    acc += g[s * H_ + h];
    gc[s * H_ + h] = acc;
  }
}

// ---------------- per-(s,h) L2-norm of q,k (in place) ----------------
__global__ __launch_bounds__(256) void qk_norm_kernel(float* __restrict__ qkvc) {
  int wid = blockIdx.x * 8 + (threadIdx.x >> 5);  // S*H*2 waves
  int lane = threadIdx.x & 31;
  if (wid >= S_ * H_ * 2) return;
  int which = wid & 1;         // 0 = q, 1 = k
  int sh = wid >> 1;
  int s = sh >> 4, h = sh & 15;
  float* p = qkvc + (size_t)s * CONV_DIM_ + which * KEY_DIM_ + h * DK_;
  float v0 = p[lane], v1 = p[lane + 32], v2 = p[lane + 64], v3 = p[lane + 96];
  float ss = v0 * v0 + v1 * v1 + v2 * v2 + v3 * v3;
#pragma unroll
  for (int m = 16; m >= 1; m >>= 1) ss += __shfl_xor(ss, m, 32);
  float inv = rsqrtf(ss + 1e-6f);
  if (which == 0) inv *= 0.08838834764831845f;  // 1/sqrt(128)
  p[lane] = v0 * inv; p[lane + 32] = v1 * inv; p[lane + 64] = v2 * inv; p[lane + 96] = v3 * inv;
}

// ---- pack bf16 operands: q, k, k*beta, q*exp(gc) in (H,S,128); k^T in (H,NC,128,64) ----
__global__ void qk_pack_kernel(const float* __restrict__ qkvc, const float* __restrict__ beta,
                               const float* __restrict__ gc,
                               bf16_t* __restrict__ qb, bf16_t* __restrict__ kb,
                               bf16_t* __restrict__ kbb, bf16_t* __restrict__ qe,
                               bf16_t* __restrict__ kT) {
  int i = blockIdx.x * 256 + threadIdx.x;     // H*S*128
  if (i >= H_ * S_ * 128) return;
  int d = i & 127;
  int s = (i >> 7) & (S_ - 1);
  int h = i >> 19;
  float q = qkvc[(size_t)s * CONV_DIM_ + h * DK_ + d];
  float k = qkvc[(size_t)s * CONV_DIM_ + KEY_DIM_ + h * DK_ + d];
  float be = beta[s * H_ + h];
  float eg = __expf(gc[s * H_ + h]);
  qb[i] = f2bf(q);
  kb[i] = f2bf(k);
  kbb[i] = f2bf(k * be);
  qe[i] = f2bf(q * eg);
  int ncc = s >> 6, cc = s & 63;
  kT[(((size_t)h * NC_ + ncc) * 128 + d) * 64 + cc] = f2bf(k);
}

// ---------------- per-(h,chunk): WMMA L & attn(bf16), then T = (I-L)^{-1} --------------
__global__ __launch_bounds__(256) void chunk_LT_kernel(
    const bf16_t* __restrict__ qb, const bf16_t* __restrict__ kb,
    const bf16_t* __restrict__ kbb, const float* __restrict__ gc,
    bf16_t* __restrict__ Tb, bf16_t* __restrict__ attn) {
  int h = blockIdx.x >> 6, nc = blockIdx.x & 63;
  int s0 = nc * CS_;
  int lane = threadIdx.x & 31, wave = threadIdx.x >> 5;
  int r = lane & 15, half = lane >> 4;
  __shared__ float sL[64][65];
  __shared__ float sT[64][65];
  __shared__ float sGc[64];
  if (threadIdx.x < 64) sGc[threadIdx.x] = gc[(s0 + threadIdx.x) * H_ + h];
  __syncthreads();

  const int isAttn = wave >> 2;            // uniform per wave
  const int mi = (wave & 3) * 16;
  const bf16_t* Abase =
      (isAttn ? qb : kbb) + ((size_t)h * S_ + s0 + mi + r) * 128 + half * 8;
  const bf16_t* Bbase = kb + ((size_t)h * S_ + s0 + r) * 128 + half * 8;

  v8f acc[4] = {};
  for (int k0 = 0; k0 < 128; k0 += 32) {
    union { v16bf v; uint4 u[2]; } av, bv;
    av.u[0] = *reinterpret_cast<const uint4*>(Abase + k0);
    av.u[1] = *reinterpret_cast<const uint4*>(Abase + k0 + 16);
#pragma unroll
    for (int t = 0; t < 4; ++t) {
      const bf16_t* pb = Bbase + (size_t)t * 16 * 128 + k0;
      bv.u[0] = *reinterpret_cast<const uint4*>(pb);
      bv.u[1] = *reinterpret_cast<const uint4*>(pb + 16);
      acc[t] = __builtin_amdgcn_wmma_f32_16x16x32_bf16(false, av.v, false, bv.v, (short)0, acc[t], false, false);
    }
  }

  size_t base = ((size_t)h * NC_ + nc) * 64 * 64;
#pragma unroll
  for (int t = 0; t < 4; ++t) {
#pragma unroll
    for (int e = 0; e < 8; ++e) {
      int i = mi + 8 * half + e;
      int j = t * 16 + r;
      float gd = sGc[i] - sGc[j];
      if (gd > 10.f) gd = 10.f;
      if (isAttn) {
        attn[base + (size_t)i * 64 + j] = f2bf((i >= j) ? acc[t][e] * __expf(gd) : 0.f);
      } else {
        sL[i][j] = (i > j) ? -acc[t][e] * __expf(gd) : 0.f;
      }
    }
  }
  __syncthreads();

  if (threadIdx.x < 64) {
    int j = threadIdx.x;
    for (int i = 0; i < 64; ++i) {
      float a2 = (i == j) ? 1.f : 0.f;
      for (int k = 0; k < i; ++k) a2 += sL[i][k] * sT[k][j];
      sT[i][j] = a2;
    }
  }
  __syncthreads();
  for (int idx = threadIdx.x; idx < 64 * 64; idx += 256)
    Tb[base + idx] = f2bf(sT[idx >> 6][idx & 63]);
}

// ------ per-(h,chunk): value(f32) = T@v_beta;  kcdn(bf16) = -(T@(k_beta*e^gc)) --------
__global__ __launch_bounds__(256) void chunk_TX_kernel(
    const float* __restrict__ qkvc, const float* __restrict__ beta,
    const float* __restrict__ gc, const bf16_t* __restrict__ Tb,
    float* __restrict__ value, bf16_t* __restrict__ kcdn) {
  int h = blockIdx.x >> 6, nc = blockIdx.x & 63;
  int s0 = nc * CS_;
  int lane = threadIdx.x & 31, wave = threadIdx.x >> 5;
  int r = lane & 15, half = lane >> 4;
  __shared__ bf16_t sXT[128][72];
  const bf16_t* Abase = Tb + ((size_t)h * NC_ + nc) * 64 * 64;

  const int mi = (wave >> 1) * 16;
  const int nj0 = (wave & 1) * 4;

  for (int pass = 0; pass < 2; ++pass) {
    __syncthreads();
    for (int idx = threadIdx.x; idx < 64 * 128; idx += 256) {
      int m = idx >> 7, d = idx & 127;
      int s = s0 + m;
      float x;
      if (pass == 0)
        x = qkvc[(size_t)s * CONV_DIM_ + 2 * KEY_DIM_ + h * DV_ + d] * beta[s * H_ + h];
      else
        x = qkvc[(size_t)s * CONV_DIM_ + KEY_DIM_ + h * DK_ + d] * beta[s * H_ + h] *
            __expf(gc[s * H_ + h]);
      sXT[d][m] = f2bf(x);
    }
    __syncthreads();

    v8f acc[4] = {};
    for (int k0 = 0; k0 < 64; k0 += 32) {
      union { v16bf v; uint4 u[2]; } av, bv;
      const bf16_t* pa = Abase + (size_t)(mi + r) * 64 + half * 8 + k0;
      av.u[0] = *reinterpret_cast<const uint4*>(pa);
      av.u[1] = *reinterpret_cast<const uint4*>(pa + 16);
#pragma unroll
      for (int t = 0; t < 4; ++t) {
        const bf16_t* pb = &sXT[(nj0 + t) * 16 + r][half * 8 + k0];
        bv.u[0] = *reinterpret_cast<const uint4*>(pb);
        bv.u[1] = *reinterpret_cast<const uint4*>(pb + 16);
        acc[t] = __builtin_amdgcn_wmma_f32_16x16x32_bf16(false, av.v, false, bv.v, (short)0, acc[t], false, false);
      }
    }
#pragma unroll
    for (int t = 0; t < 4; ++t) {
      int d0 = (nj0 + t) * 16 + r;
#pragma unroll
      for (int e = 0; e < 8; ++e) {
        int c = mi + 8 * half + e;
        size_t o = ((size_t)h * S_ + s0 + c) * 128 + d0;
        if (pass == 0) value[o] = acc[t][e];
        else           kcdn[o] = f2bf(-acc[t][e]);
      }
    }
  }
}

// -------- WMMA inter-chunk scan; grid = H * 8 (DV tiles of 16), 128 thr = 4 waves -----
// f32 state master in LDS; bf16 transposed copies feed WMMA B operands.
__global__ __launch_bounds__(128) void scan_kernel(
    const bf16_t* __restrict__ qe, const bf16_t* __restrict__ kcdn,
    const bf16_t* __restrict__ kT, const bf16_t* __restrict__ attn,
    const float* __restrict__ value, const float* __restrict__ gc,
    float* __restrict__ core, float* __restrict__ rec) {
  int h = blockIdx.x >> 3;
  int vt = (blockIdx.x & 7) * 16;
  int t = threadIdx.x;
  int lane = t & 31, wave = t >> 5;
  int r = lane & 15, half = lane >> 4;
  __shared__ float  sS[128][17];     // state[d][vl] (f32 master)
  __shared__ bf16_t sSbT[16][136];   // state^T bf16 (B operand), 272B rows
  __shared__ bf16_t sVnT[16][72];    // v_new^T bf16
  __shared__ bf16_t sVnTg[16][72];   // (v_new * e^{glast-gc})^T bf16
  __shared__ float  sEd[64];

  const bf16_t bz = __builtin_bit_cast(__bf16, (unsigned short)0);
  for (int idx = t; idx < 128 * 16; idx += 128) sS[idx >> 4][idx & 15] = 0.f;
  for (int idx = t; idx < 16 * 128; idx += 128) sSbT[idx >> 7][idx & 127] = bz;

  const int mi = wave * 16;          // M strip for steps 1-2

  for (int nc = 0; nc < NC_; ++nc) {
    int s0 = nc * CS_;
    float glast = gc[(s0 + 63) * H_ + h];
    if (t < 64) sEd[t] = __expf(glast - gc[(s0 + t) * H_ + h]);
    __syncthreads();                 // sSbT + sEd ready

    union { v16bf v; uint4 u[2]; } av, bv;

    // ---- step 1: v_new = value - kcd @ state  (A = -kcd, C = value)
    v8f acc;
#pragma unroll
    for (int e = 0; e < 8; ++e)
      acc[e] = value[((size_t)h * S_ + s0 + mi + 8 * half + e) * 128 + vt + r];
#pragma unroll
    for (int k0 = 0; k0 < 128; k0 += 32) {
      const bf16_t* pa = kcdn + ((size_t)h * S_ + s0 + mi + r) * 128 + half * 8 + k0;
      av.u[0] = *reinterpret_cast<const uint4*>(pa);
      av.u[1] = *reinterpret_cast<const uint4*>(pa + 16);
      const bf16_t* pb = &sSbT[r][half * 8 + k0];
      bv.u[0] = *reinterpret_cast<const uint4*>(pb);
      bv.u[1] = *reinterpret_cast<const uint4*>(pb + 16);
      acc = __builtin_amdgcn_wmma_f32_16x16x32_bf16(false, av.v, false, bv.v, (short)0, acc, false, false);
    }
#pragma unroll
    for (int e = 0; e < 8; ++e) {
      int c = mi + 8 * half + e;
      float vn = acc[e];
      sVnT[r][c] = f2bf(vn);
      sVnTg[r][c] = f2bf(vn * sEd[c]);
    }
    __syncthreads();

    // ---- step 2: out = (q*e^gc) @ state + attn @ v_new
    v8f acc2 = {};
#pragma unroll
    for (int k0 = 0; k0 < 128; k0 += 32) {
      const bf16_t* pa = qe + ((size_t)h * S_ + s0 + mi + r) * 128 + half * 8 + k0;
      av.u[0] = *reinterpret_cast<const uint4*>(pa);
      av.u[1] = *reinterpret_cast<const uint4*>(pa + 16);
      const bf16_t* pb = &sSbT[r][half * 8 + k0];
      bv.u[0] = *reinterpret_cast<const uint4*>(pb);
      bv.u[1] = *reinterpret_cast<const uint4*>(pb + 16);
      acc2 = __builtin_amdgcn_wmma_f32_16x16x32_bf16(false, av.v, false, bv.v, (short)0, acc2, false, false);
    }
#pragma unroll
    for (int k0 = 0; k0 < 64; k0 += 32) {
      const bf16_t* pa = attn + (((size_t)h * NC_ + nc) * 64 + mi + r) * 64 + half * 8 + k0;
      av.u[0] = *reinterpret_cast<const uint4*>(pa);
      av.u[1] = *reinterpret_cast<const uint4*>(pa + 16);
      const bf16_t* pb = &sVnT[r][half * 8 + k0];
      bv.u[0] = *reinterpret_cast<const uint4*>(pb);
      bv.u[1] = *reinterpret_cast<const uint4*>(pb + 16);
      acc2 = __builtin_amdgcn_wmma_f32_16x16x32_bf16(false, av.v, false, bv.v, (short)0, acc2, false, false);
    }
#pragma unroll
    for (int e = 0; e < 8; ++e)
      core[((size_t)(s0 + mi + 8 * half + e) * H_ + h) * 128 + vt + r] = acc2[e];

    // ---- step 3: state = state*e^glast + k^T @ (v_new scaled)
    float egl = __expf(glast);
#pragma unroll
    for (int mt = 0; mt < 2; ++mt) {
      int db = (wave * 2 + mt) * 16;
      v8f acc3;
#pragma unroll
      for (int e = 0; e < 8; ++e) acc3[e] = sS[db + 8 * half + e][r] * egl;
#pragma unroll
      for (int k0 = 0; k0 < 64; k0 += 32) {
        const bf16_t* pa = kT + (((size_t)h * NC_ + nc) * 128 + db + r) * 64 + half * 8 + k0;
        av.u[0] = *reinterpret_cast<const uint4*>(pa);
        av.u[1] = *reinterpret_cast<const uint4*>(pa + 16);
        const bf16_t* pb = &sVnTg[r][half * 8 + k0];
        bv.u[0] = *reinterpret_cast<const uint4*>(pb);
        bv.u[1] = *reinterpret_cast<const uint4*>(pb + 16);
        acc3 = __builtin_amdgcn_wmma_f32_16x16x32_bf16(false, av.v, false, bv.v, (short)0, acc3, false, false);
      }
#pragma unroll
      for (int e = 0; e < 8; ++e) sS[db + 8 * half + e][r] = acc3[e];
    }
    __syncthreads();
    // rebuild transposed bf16 state for next chunk
    for (int idx = t; idx < 16 * 128; idx += 128) {
      int vl = idx >> 7, d = idx & 127;
      sSbT[vl][d] = f2bf(sS[d][vl]);
    }
  }
  __syncthreads();
  for (int idx = t; idx < 128 * 16; idx += 128)
    rec[((size_t)h * 128 + (idx >> 4)) * 128 + vt + (idx & 15)] = sS[idx >> 4][idx & 15];
}

// ---------------- RMS-norm * norm_w * silu(z)  ->  bf16 ----------------
__global__ __launch_bounds__(256) void gate_norm_kernel(
    const float* __restrict__ core, const float* __restrict__ z,
    const float* __restrict__ nw, bf16_t* __restrict__ outg) {
  int wid = blockIdx.x * 8 + (threadIdx.x >> 5);  // S*H waves
  int lane = threadIdx.x & 31;
  if (wid >= S_ * H_) return;
  int s = wid >> 4, h = wid & 15;
  const float* row = core + (size_t)wid * 128;
  float v0 = row[lane], v1 = row[lane + 32], v2 = row[lane + 64], v3 = row[lane + 96];
  float ss = v0 * v0 + v1 * v1 + v2 * v2 + v3 * v3;
#pragma unroll
  for (int m = 16; m >= 1; m >>= 1) ss += __shfl_xor(ss, m, 32);
  float inv = rsqrtf(ss * (1.f / 128.f) + 1e-6f);
  const float* zr = z + (size_t)s * VAL_DIM_ + h * DV_;
  bf16_t* orow = outg + (size_t)s * VAL_DIM_ + h * DV_;
  orow[lane]      = f2bf(nw[lane]      * v0 * inv * siluf_(zr[lane]));
  orow[lane + 32] = f2bf(nw[lane + 32] * v1 * inv * siluf_(zr[lane + 32]));
  orow[lane + 64] = f2bf(nw[lane + 64] * v2 * inv * siluf_(zr[lane + 64]));
  orow[lane + 96] = f2bf(nw[lane + 96] * v3 * inv * siluf_(zr[lane + 96]));
}

// ---------------- host launcher ----------------
extern "C" void kernel_launch(void* const* d_in, const int* in_sizes, int n_in,
                              void* d_out, int out_size, void* d_ws, size_t ws_size,
                              hipStream_t stream) {
  const float* hidden = (const float*)d_in[0];
  const float* conv_w = (const float*)d_in[1];
  const float* qkv_w  = (const float*)d_in[2];
  const float* z_w    = (const float*)d_in[3];
  const float* b_w    = (const float*)d_in[4];
  const float* a_w    = (const float*)d_in[5];
  const float* out_w  = (const float*)d_in[6];
  const float* dt_b   = (const float*)d_in[7];
  const float* A_log  = (const float*)d_in[8];
  const float* norm_w = (const float*)d_in[9];
  float* out = (float*)d_out;

  size_t off = 0;
  auto alloc = [&](size_t bytes) {
    size_t r = off;
    off = (off + bytes + 255) & ~(size_t)255;
    return (char*)d_ws + r;
  };
  bf16_t* hbf    = (bf16_t*)alloc((size_t)S_ * HID_ * 2);
  bf16_t* wq     = (bf16_t*)alloc((size_t)CONV_DIM_ * HID_ * 2);
  bf16_t* wz     = (bf16_t*)alloc((size_t)VAL_DIM_ * HID_ * 2);
  bf16_t* wo     = (bf16_t*)alloc((size_t)HID_ * VAL_DIM_ * 2);
  float*  mixed  = (float*)alloc((size_t)S_ * CONV_DIM_ * 4);
  float*  qkvc   = (float*)alloc((size_t)S_ * CONV_DIM_ * 4);
  float*  zb     = (float*)alloc((size_t)S_ * VAL_DIM_ * 4);
  float*  ab     = (float*)alloc((size_t)S_ * 32 * 4);
  float*  betab  = (float*)alloc((size_t)S_ * H_ * 4);
  float*  gb     = (float*)alloc((size_t)S_ * H_ * 4);
  float*  gcb    = (float*)alloc((size_t)S_ * H_ * 4);
  bf16_t* qbb    = (bf16_t*)alloc((size_t)H_ * S_ * 128 * 2);
  bf16_t* kbb2   = (bf16_t*)alloc((size_t)H_ * S_ * 128 * 2);
  bf16_t* kbetab = (bf16_t*)alloc((size_t)H_ * S_ * 128 * 2);
  bf16_t* qeb    = (bf16_t*)alloc((size_t)H_ * S_ * 128 * 2);
  bf16_t* kTb    = (bf16_t*)alloc((size_t)H_ * NC_ * 128 * 64 * 2);
  bf16_t* Tbb    = (bf16_t*)alloc((size_t)H_ * NC_ * 64 * 64 * 2);
  bf16_t* attnbf = (bf16_t*)alloc((size_t)H_ * NC_ * 64 * 64 * 2);
  float*  valueb = (float*)alloc((size_t)H_ * S_ * 128 * 4);
  bf16_t* kcdnb  = (bf16_t*)alloc((size_t)H_ * S_ * 128 * 2);
  float*  coreb  = (float*)alloc((size_t)S_ * H_ * 128 * 4);
  bf16_t* outgbf = (bf16_t*)alloc((size_t)S_ * VAL_DIM_ * 2);

  float* out_hidden = out;                                   // S*HID
  float* out_conv   = out + (size_t)S_ * HID_;               // CONV_DIM*K
  float* out_rec    = out_conv + (size_t)CONV_DIM_ * Kc_;    // H*DK*DV

  // 1) casts to bf16
  {
    int n = S_ * HID_;
    cast_bf16_kernel<<<(n + 255) / 256, 256, 0, stream>>>(hidden, hbf, n);
    n = CONV_DIM_ * HID_;
    cast_bf16_kernel<<<(n + 255) / 256, 256, 0, stream>>>(qkv_w, wq, n);
    n = VAL_DIM_ * HID_;
    cast_bf16_kernel<<<(n + 255) / 256, 256, 0, stream>>>(z_w, wz, n);
    n = HID_ * VAL_DIM_;
    cast_bf16_kernel<<<(n + 255) / 256, 256, 0, stream>>>(out_w, wo, n);
  }

  // 2) big projections on WMMA
  gemm_bf16_kernel<<<dim3(CONV_DIM_ / 256, S_ / 32), 256, 0, stream>>>(
      hbf, wq, mixed, S_, CONV_DIM_, HID_);
  gemm_bf16_kernel<<<dim3(VAL_DIM_ / 256, S_ / 32), 256, 0, stream>>>(
      hbf, wz, zb, S_, VAL_DIM_, HID_);

  // 3) tiny b/a projections
  ab_proj_kernel<<<(S_ * 32 + 127) / 128, 128, 0, stream>>>(hidden, b_w, a_w, ab);

  // 4) conv_state (pre-conv), then depthwise conv + silu
  conv_state_kernel<<<(CONV_DIM_ * Kc_ + 255) / 256, 256, 0, stream>>>(mixed, out_conv);
  conv_silu_kernel<<<(S_ * CONV_DIM_ + 255) / 256, 256, 0, stream>>>(mixed, conv_w, qkvc);

  // 5) gates + per-chunk cumsum
  gates_kernel<<<(S_ * H_ + 255) / 256, 256, 0, stream>>>(ab, A_log, dt_b, betab, gb);
  cumsum_g_kernel<<<(H_ * NC_ + 255) / 256, 256, 0, stream>>>(gb, gcb);

  // 6) q/k L2-norm in place, then bf16 head-major packs
  qk_norm_kernel<<<(S_ * H_ * 2) / 8, 256, 0, stream>>>(qkvc);
  qk_pack_kernel<<<(H_ * S_ * 128 + 255) / 256, 256, 0, stream>>>(
      qkvc, betab, gcb, qbb, kbb2, kbetab, qeb, kTb);

  // 7) per-chunk WMMA: L/attn + triangular inverse, then value & -k_cumdecay
  chunk_LT_kernel<<<H_ * NC_, 256, 0, stream>>>(qbb, kbb2, kbetab, gcb, Tbb, attnbf);
  chunk_TX_kernel<<<H_ * NC_, 256, 0, stream>>>(qkvc, betab, gcb, Tbb, valueb, kcdnb);

  // 8) WMMA inter-chunk scan (parallel over head x DV-tile of 16)
  scan_kernel<<<H_ * 8, 128, 0, stream>>>(qeb, kcdnb, kTb, attnbf, valueb, gcb,
                                          coreb, out_rec);

  // 9) RMS-norm gate -> bf16, final out-projection on WMMA
  gate_norm_kernel<<<(S_ * H_) / 8, 256, 0, stream>>>(coreb, zb, norm_w, outgbf);
  gemm_bf16_kernel<<<dim3(HID_ / 256, S_ / 32), 256, 0, stream>>>(
      outgbf, wo, out_hidden, S_, HID_, VAL_DIM_);
}
